// Model_9242769622541
// MI455X (gfx1250) — matrix operand order repaired and checked by reading, not compiled, not run on hardware
//
#include <hip/hip_runtime.h>
#include <hip/hip_bf16.h>
#include <math.h>

// ---------------------------------------------------------------------------
// RAD-TTS alignment attention for MI455X (gfx1250, wave32, WMMA).
// Dominant GEMM (key conv1: 1024x1536x256 per batch, x8) and the Q.K einsum
// run on v_wmma_f32_16x16x32_bf16 (fp32 accumulate). Weights are pre-shuffled
// into WMMA fragment-major order -> each A fragment is 2x global_load_b128
// with immediate offsets; A loads are double-buffered so WMMA overlaps VMEM.
// ---------------------------------------------------------------------------

typedef __attribute__((ext_vector_type(16))) __bf16 v16bf;
typedef __attribute__((ext_vector_type(8)))  float  v8f;

struct __align__(16) Frag8 { unsigned int u[8]; };
union FragB16 { Frag8 f; v16bf v; };

__device__ __forceinline__ unsigned short f2bf(float f) {
  unsigned int u = __float_as_uint(f);
  u += 0x7FFFu + ((u >> 16) & 1u);          // round-to-nearest-even
  return (unsigned short)(u >> 16);
}

__device__ __forceinline__ v8f zero8() {
  v8f z = {0.f, 0.f, 0.f, 0.f, 0.f, 0.f, 0.f, 0.f};
  return z;
}

#define ALIGN_TEMP 0.0005f
// byte stride between the 8 M-tiles a wave owns in a1f (8 tiles apart)
#define A1_TILE_STRIDE (8 * 48 * 32 * 32)   // 0x60000, fits 24-bit imm offset

// ---------------------------------------------------------------------------
// Shuffle fp32 weights [Mtiles*16][rowK] into WMMA A-fragment-major bf16:
// dst[((mt*Ksteps + ks)*32 + lane)*16 + e], in contiguous K order, not the ISA table order
// (lane = h*16 + m%16; element e -> kk = h*16 + e within the 32-wide K step).
// ---------------------------------------------------------------------------
__global__ void shuffle_fragA(const float* __restrict__ src,
                              unsigned short* __restrict__ dst,
                              int Mtiles, int Ksteps, int rowK) {
  int n = Mtiles * Ksteps * 512;
  for (int idx = blockIdx.x * blockDim.x + threadIdx.x; idx < n;
       idx += gridDim.x * blockDim.x) {
    int e    = idx & 15;
    int lane = (idx >> 4) & 31;
    int rest = idx >> 9;
    int ks   = rest % Ksteps;
    int mt   = rest / Ksteps;
    int m = mt * 16 + (lane & 15);
    int h = lane >> 4;
    int j = e >> 1, p = e & 1;
    int kk = h * 16 + 2 * j + p;
    dst[idx] = f2bf(src[m * rowK + ks * 32 + kk]);
  }
}

// ---------------------------------------------------------------------------
// Key encoder: keys = conv1x1(relu(conv3(phonemes)))   [8][80][256]
// One workgroup per (batch, 16 Tk columns). 8 waves.
// GEMM1: M=1024 K=1536 N=16 (im2col in LDS, two K-phases of 768).
// GEMM2: M=80   K=1024 N=16 (hidden tile in LDS).
// ---------------------------------------------------------------------------
__global__ __launch_bounds__(256) void keys_kernel(
    const float* __restrict__ phon,          // [8][512][256]
    const float* __restrict__ kb1,           // [1024]
    const float* __restrict__ kb2,           // [80]
    const unsigned short* __restrict__ a1f,  // frag-major [64][48][32][16]
    const unsigned short* __restrict__ a2f,  // frag-major [5][32][32][16]
    float* __restrict__ keysF)               // [8][80][256]
{
  const int b    = blockIdx.y;
  const int t0   = blockIdx.x * 16;
  const int tid  = threadIdx.x;
  const int lane = tid & 31;
  const int w    = tid >> 5;
  const int h    = lane >> 4;
  const int ln   = lane & 15;

  __shared__ __align__(16) unsigned short sB[16 * 776];   // im2col [n][k<768]
  __shared__ __align__(16) unsigned short sH[16 * 1032];  // hidden [n][m<1024]

  v8f acc[8];
#pragma unroll
  for (int i = 0; i < 8; ++i) acc[i] = zero8();

  const unsigned short* sBrow = &sB[ln * 776 + h * 16];

  for (int p = 0; p < 2; ++p) {
    // im2col fill: k_global = p*768 + kl = cin*3 + tap, 'SAME' zero pad.
    for (int f = tid; f < 768 * 16; f += 256) {
      int kl = f >> 4, n = f & 15;
      int kg = p * 768 + kl;
      int ci = kg / 3, tap = kg - ci * 3;
      int t  = t0 + n + tap - 1;
      float v = (t >= 0 && t < 256) ? phon[(b * 512 + ci) * 256 + t] : 0.0f;
      sB[n * 776 + kl] = f2bf(v);
    }
    __syncthreads();
    // wave base pointer into frag-major weights; +1024 B per K-step
    const char* pw = (const char*)a1f +
                     (((size_t)w * 48 + (size_t)p * 24) * 32 + lane) * 32;
    for (int ks = 0; ks < 24; ++ks) {
      FragB16 Bf;
      Bf.f = *(const Frag8*)(sBrow + ks * 32);
      __builtin_prefetch(pw + 1024, 0, 0);  // next K-step of weight stream
      FragB16 A0, A1;
      A0.f = *(const Frag8*)pw;
#pragma unroll
      for (int i = 0; i < 8; ++i) {
        if (i < 7)  // double-buffer: issue next tile's loads before this WMMA
          ((i & 1) ? A0 : A1).f =
              *(const Frag8*)(pw + (i + 1) * A1_TILE_STRIDE);
        acc[i] = __builtin_amdgcn_wmma_f32_16x16x32_bf16(
            false, ((i & 1) ? A1 : A0).v, false, Bf.v, (short)0, acc[i],
            false, false);
      }
      pw += 1024;
    }
    __syncthreads();
  }

  // bias + ReLU -> bf16 hidden tile in B-matrix layout: sH[n][m]
#pragma unroll
  for (int i = 0; i < 8; ++i) {
    const int Mb = (i * 8 + w) * 16;
    unsigned int* hp = (unsigned int*)&sH[ln * 1032 + Mb + 8 * h];
#pragma unroll
    for (int q = 0; q < 4; ++q) {
      int m0 = Mb + 8 * h + 2 * q;
      float v0 = fmaxf(acc[i][2 * q]     + kb1[m0],     0.0f);
      float v1 = fmaxf(acc[i][2 * q + 1] + kb1[m0 + 1], 0.0f);
      hp[q] = (unsigned int)f2bf(v0) | ((unsigned int)f2bf(v1) << 16);
    }
  }
  __syncthreads();

  if (w < 5) {  // wave-uniform: 5 waves cover M=80
    v8f acc2 = zero8();
    const char* pw2 = (const char*)a2f + ((size_t)(w * 32) * 32 + lane) * 32;
    const unsigned short* sHrow = &sH[ln * 1032 + h * 16];
    FragB16 A0, A1;
    A0.f = *(const Frag8*)pw2;
#pragma unroll
    for (int ks = 0; ks < 32; ++ks) {
      if (ks < 31)
        ((ks & 1) ? A0 : A1).f = *(const Frag8*)(pw2 + (ks + 1) * 1024);
      FragB16 Bf;
      Bf.f = *(const Frag8*)(sHrow + ks * 32);
      acc2 = __builtin_amdgcn_wmma_f32_16x16x32_bf16(
          false, ((ks & 1) ? A1 : A0).v, false, Bf.v, (short)0, acc2, false,
          false);
    }
#pragma unroll
    for (int r = 0; r < 8; ++r) {
      int mo = w * 16 + 8 * h + r;
      keysF[(b * 80 + mo) * 256 + t0 + ln] = acc2[r] + kb2[mo];
    }
  }
}

// ---------------------------------------------------------------------------
// Query encoder (small: ~0.9 GFLOP total) — LDS-tiled VALU.
// ---------------------------------------------------------------------------
__global__ __launch_bounds__(256) void queries_kernel(
    const float* __restrict__ audio,  // [8][80][1000]
    const float* __restrict__ qw1, const float* __restrict__ qb1,
    const float* __restrict__ qw2, const float* __restrict__ qb2,
    const float* __restrict__ qw3, const float* __restrict__ qb3,
    float* __restrict__ qF)           // [8][80][1000]
{
  const int b   = blockIdx.y;
  const int t0  = blockIdx.x * 16;
  const int tid = threadIdx.x;
  __shared__ float xa[80 * 18];
  __shared__ float h1[160 * 16];
  __shared__ float h2[80 * 16];

  for (int f = tid; f < 80 * 18; f += 256) {
    int ci = f / 18, j = f - ci * 18;
    int t  = t0 + j - 1;
    xa[f] = (t >= 0 && t < 1000) ? audio[(b * 80 + ci) * 1000 + t] : 0.0f;
  }
  __syncthreads();
  for (int f = tid; f < 160 * 16; f += 256) {
    int c = f >> 4, n = f & 15;
    float acc = qb1[c];
    const float* wp = qw1 + c * 240;
    for (int ci = 0; ci < 80; ++ci)
      acc += wp[ci * 3 + 0] * xa[ci * 18 + n]
           + wp[ci * 3 + 1] * xa[ci * 18 + n + 1]
           + wp[ci * 3 + 2] * xa[ci * 18 + n + 2];
    h1[f] = fmaxf(acc, 0.0f);
  }
  __syncthreads();
  for (int f = tid; f < 80 * 16; f += 256) {
    int c = f >> 4, n = f & 15;
    float acc = qb2[c];
    const float* wp = qw2 + c * 160;
    for (int ci = 0; ci < 160; ++ci) acc += wp[ci] * h1[ci * 16 + n];
    h2[f] = fmaxf(acc, 0.0f);
  }
  __syncthreads();
  for (int f = tid; f < 80 * 16; f += 256) {
    int c = f >> 4, n = f & 15;
    float acc = qb3[c];
    const float* wp = qw3 + c * 80;
    for (int ci = 0; ci < 80; ++ci) acc += wp[ci] * h2[ci * 16 + n];
    int t = t0 + n;
    if (t < 1000) qF[(b * 80 + c) * 1000 + t] = acc;
  }
}

// ---------------------------------------------------------------------------
// Attention: WMMA Q.K (M=16 tq, N=256 tk in two 128 passes, K=80 pad 96),
// then per-row log-softmax over Tk=256, + log(prior+1e-8), mask -> -inf.
// Q tile stored in LDS in fragment-major order -> 2x ds_load_b128 per frag.
// ---------------------------------------------------------------------------
__global__ __launch_bounds__(256) void attn_kernel(
    const float* __restrict__ qF,             // [8][80][1000]
    const float* __restrict__ keysF,          // [8][80][256]
    const float* __restrict__ prior,          // [8][1000][256]
    const unsigned char* __restrict__ maskp,  // [8][1000][256] bool
    float* __restrict__ out)                  // [8][1000][256]
{
  const int b    = blockIdx.y;
  const int tq0  = blockIdx.x * 16;
  const int tid  = threadIdx.x;
  const int lane = tid & 31;
  const int w    = tid >> 5;
  const int h    = lane >> 4;
  const int ln   = lane & 15;

  __shared__ __align__(16) unsigned short sAf[3 * 512];   // Q frag-major
  __shared__ __align__(16) unsigned short sK[128 * 104];  // K [n][c<96]
  __shared__ float sS[16 * 256];
  __shared__ float sQ2[16];
  __shared__ float sK2[256];

  for (int f = tid; f < 16 * 96; f += 256) {
    int m = f / 96, c = f - m * 96;
    int tq = tq0 + m;
    float v = (c < 80 && tq < 1000) ? qF[(b * 80 + c) * 1000 + tq] : 0.0f;
    // A-fragment contiguous layout: c = ks*32 + kk, kk = hh*16 + e
    int ks = c >> 5, kk = c & 31;
    int hh = kk >> 4;
    int e  = kk & 15;
    sAf[ks * 512 + (hh * 16 + m) * 16 + e] = f2bf(v);
  }
  if (tid < 16) {
    int tq = tq0 + tid;
    float s = 0.0f;
    if (tq < 1000)
      for (int c = 0; c < 80; ++c) {
        float v = qF[(b * 80 + c) * 1000 + tq];
        s += v * v;
      }
    sQ2[tid] = s;
  }
  {
    float s = 0.0f;
    for (int c = 0; c < 80; ++c) {
      float v = keysF[(b * 80 + c) * 256 + tid];
      s += v * v;
    }
    sK2[tid] = s;
  }

  for (int hp = 0; hp < 2; ++hp) {
    __syncthreads();
    for (int f = tid; f < 128 * 96; f += 256) {
      int nl = f / 96, c = f - nl * 96;
      int n  = hp * 128 + nl;
      float v = (c < 80) ? keysF[(b * 80 + c) * 256 + n] : 0.0f;
      sK[nl * 104 + c] = f2bf(v);
    }
    __syncthreads();

    v8f acc = zero8();
#pragma unroll
    for (int ks = 0; ks < 3; ++ks) {
      FragB16 Af, Bf;
      Af.f = *(const Frag8*)&sAf[ks * 512 + lane * 16];
      Bf.f = *(const Frag8*)&sK[(w * 16 + ln) * 104 + ks * 32 + h * 16];
      acc = __builtin_amdgcn_wmma_f32_16x16x32_bf16(
          false, Af.v, false, Bf.v, (short)0, acc, false, false);
    }
    const int n = hp * 128 + w * 16 + ln;
#pragma unroll
    for (int r = 0; r < 8; ++r) {
      int m = 8 * h + r;
      sS[m * 256 + n] = -ALIGN_TEMP * (sQ2[m] + sK2[n] - 2.0f * acc[r]);
    }
  }
  __syncthreads();

  // wave32 shuffle-reduced log-softmax; wave w owns rows 2w, 2w+1
  for (int rr = 0; rr < 2; ++rr) {
    const int row = w * 2 + rr;
    const int tq  = tq0 + row;
    float mx = -__builtin_inff();
    for (int i = 0; i < 8; ++i)
      mx = fmaxf(mx, sS[row * 256 + lane + 32 * i]);
#pragma unroll
    for (int off = 16; off > 0; off >>= 1)
      mx = fmaxf(mx, __shfl_xor(mx, off, 32));
    float se = 0.0f;
    for (int i = 0; i < 8; ++i)
      se += expf(sS[row * 256 + lane + 32 * i] - mx);
#pragma unroll
    for (int off = 16; off > 0; off >>= 1)
      se += __shfl_xor(se, off, 32);
    const float lse = mx + logf(se);
    if (tq < 1000) {
      for (int i = 0; i < 8; ++i) {
        int n = lane + 32 * i;
        size_t idx = ((size_t)(b * 1000 + tq)) * 256 + (size_t)n;
        float v = sS[row * 256 + n] - lse + logf(prior[idx] + 1e-8f);
        if (!maskp[idx]) v = -__builtin_inff();
        out[idx] = v;
      }
    }
  }
}

// ---------------------------------------------------------------------------
extern "C" void kernel_launch(void* const* d_in, const int* in_sizes, int n_in,
                              void* d_out, int out_size, void* d_ws,
                              size_t ws_size, hipStream_t stream) {
  (void)in_sizes; (void)n_in; (void)out_size; (void)ws_size;
  const float* phon  = (const float*)d_in[0];
  const float* audio = (const float*)d_in[1];
  const unsigned char* mask = (const unsigned char*)d_in[2];
  const float* prior = (const float*)d_in[3];
  const float* kw1 = (const float*)d_in[4];
  const float* kb1 = (const float*)d_in[5];
  const float* kw2 = (const float*)d_in[6];
  const float* kb2 = (const float*)d_in[7];
  const float* qw1 = (const float*)d_in[8];
  const float* qb1 = (const float*)d_in[9];
  const float* qw2 = (const float*)d_in[10];
  const float* qb2 = (const float*)d_in[11];
  const float* qw3 = (const float*)d_in[12];
  const float* qb3 = (const float*)d_in[13];
  float* outF = (float*)d_out;

  // Workspace layout (bytes):
  //   a1f  : 64*48*32*16 bf16 = 3,145,728
  //   a2f  : 5*32*32*16  bf16 =   163,840
  //   keysF: 8*80*256  f32    =   655,360
  //   qF   : 8*80*1000 f32    = 2,560,000     total ~6.5 MB
  char* ws = (char*)d_ws;
  unsigned short* a1f = (unsigned short*)ws;
  unsigned short* a2f = a1f + 64 * 48 * 512;
  float* keysF = (float*)(ws + 3145728 + 163840);
  float* qF    = keysF + 8 * 80 * 256;

  shuffle_fragA<<<dim3(6144), dim3(256), 0, stream>>>(kw1, a1f, 64, 48, 1536);
  shuffle_fragA<<<dim3(320),  dim3(256), 0, stream>>>(kw2, a2f, 5, 32, 1024);
  keys_kernel<<<dim3(16, 8), dim3(256), 0, stream>>>(phon, kb1, kb2, a1f, a2f,
                                                     keysF);
  queries_kernel<<<dim3(63, 8), dim3(256), 0, stream>>>(audio, qw1, qb1, qw2,
                                                        qb2, qw3, qb3, qF);
  attn_kernel<<<dim3(63, 8), dim3(256), 0, stream>>>(qF, keysF, prior, mask,
                                                     outF);
}
